// CustomNeighborhoodAttention2D_65678639891242
// MI455X (gfx1250) — matrix operand order repaired and checked by reading, not compiled, not run on hardware
//
#include <hip/hip_runtime.h>
#include <hip/hip_bf16.h>
#include <stdint.h>

typedef __attribute__((ext_vector_type(16))) _Float16 v16h;
typedef __attribute__((ext_vector_type(8)))  float    v8f;

#define NA_H 64
#define NA_W 128
#define NA_C 256
#define NA_NH 8
#define NA_HD 32
#define NA_KK 7
#define NA_WIN 49
#define NA_PIX (NA_H * NA_W)   // 8192
#define GEMM_K 256             // K dim of both GEMMs (== C), compile-time
#define KSTEPS (GEMM_K / 32)   // 8

// ---------------------------------------------------------------------------
// fp32 -> f16 conversion (straight copy)
// ---------------------------------------------------------------------------
__global__ void na_cvt_f16(const float* __restrict__ in, _Float16* __restrict__ out, int n) {
    int i = blockIdx.x * blockDim.x + threadIdx.x;
    if (i < n) out[i] = (_Float16)in[i];
}

// fp32 [K,N] -> f16 [N,K] transpose-convert (weights; coalesced reads)
__global__ void na_cvt_f16_t(const float* __restrict__ in, _Float16* __restrict__ out,
                             int K, int N) {
    int i = blockIdx.x * blockDim.x + threadIdx.x;
    if (i < K * N) {
        int k = i / N, n = i - k * N;
        out[(size_t)n * K + k] = (_Float16)in[i];
    }
}

// ---------------------------------------------------------------------------
// WMMA GEMM: C[M,N] = A[M,256] (f16 row-major) * BT[N,256] (f16 transposed) + bias
// One wave32 per 16x64 output strip; K fully unrolled (8 steps, 32 v_wmma).
// Double-buffered fragments: step s+1's 10 b128 loads are issued before step
// s's 4 wmmas, so wmmas retire behind partial loadcnt waits instead of 0x0.
// Fragment placement (ISA 7.12.2, 16-bit, wave32):
//   A 16x32: lane 0-15 M=lane; VGPR v K-pair k0 = (v/4)*16 + (v%4)*2 + (lane>=16)*8
//   B 32x16: N = lane&15; VGPR v K rows (v/4)*16 + (v%4)*2 + (lane>=16)*8, +1
//   C 16x16 f32: VGPR r -> M = r + (lane>=16)*8, N = lane&15
// ---------------------------------------------------------------------------
union NaFrag { v16h h; uint32_t u[8]; };

__device__ __forceinline__ void na_load_a(NaFrag& a, const _Float16* Ap, int kk, int hi) {
#pragma unroll
    for (int v = 0; v < 8; ++v) {
        const int k0 = ((v >> 2) << 4) + ((v & 3) << 1) + (hi << 3);
        a.u[v] = *(const uint32_t*)(Ap + kk + k0);       // 2 f16 along K
    }
}

__device__ __forceinline__ void na_load_b4(NaFrag* b, const _Float16* B0, size_t bts,
                                           int kk, int hi) {
#pragma unroll
    for (int t = 0; t < 4; ++t)
#pragma unroll
        for (int v = 0; v < 8; ++v) {
            const int kr = ((v >> 2) << 4) + ((v & 3) << 1) + (hi << 3);
            b[t].u[v] = *(const uint32_t*)(B0 + t * bts + kk + kr);
        }
}

__global__ void __launch_bounds__(256, 1)
na_wmma_gemm(const _Float16* __restrict__ A, const _Float16* __restrict__ BT,
             const float* __restrict__ bias, float* __restrict__ C,
             int M, int N) {
    const int wave = (blockIdx.x * blockDim.x + threadIdx.x) >> 5;
    const int lane = threadIdx.x & 31;
    const int stripsN = N >> 6;                // 64-wide strips
    const int tm = wave / stripsN;
    const int tn = wave - tm * stripsN;
    if (tm * 16 >= M) return;                  // wave-uniform; EXEC stays all-ones

    const int col = lane & 15;                 // A row-in-tile / B,C column
    const int hi  = lane >> 4;                 // lane-group select

    const _Float16* Ap = A + (size_t)(tm * 16 + col) * GEMM_K;
    const _Float16* B0 = BT + (size_t)(tn * 64 + col) * GEMM_K; // strip sub-tile 0
    const size_t bts = (size_t)16 * GEMM_K;                     // 16 columns of BT

    v8f acc[4] = {{}, {}, {}, {}};

    NaFrag a[2], b[2][4];
    na_load_a (a[0],     Ap, 0, hi);
    na_load_b4(b[0], B0, bts, 0, hi);

#pragma unroll
    for (int s = 0; s < KSTEPS; ++s) {
        const int cur = s & 1;
        const int nxt = cur ^ 1;
        if (s + 1 < KSTEPS) {                  // issue next step's loads first
            na_load_a (a[nxt],     Ap, (s + 1) * 32, hi);
            na_load_b4(b[nxt], B0, bts, (s + 1) * 32, hi);
        }
#pragma unroll
        for (int t = 0; t < 4; ++t) {
            acc[t] = __builtin_amdgcn_wmma_f32_16x16x32_f16(
                         /*neg_a=*/false, a[cur].h, /*neg_b=*/false, b[cur][t].h,
                         /*c_mod=*/(short)0, acc[t],
                         /*reuse_a=*/false, /*reuse_b=*/false);
        }
    }

#pragma unroll
    for (int t = 0; t < 4; ++t) {
        const int cc = tn * 64 + t * 16 + col;
        const float bv = bias[cc];
#pragma unroll
        for (int r = 0; r < 8; ++r) {
            const int rr = tm * 16 + r + hi * 8;
            C[(size_t)rr * N + cc] = acc[t][r] + bv;
        }
    }
}

// ---------------------------------------------------------------------------
// Neighborhood attention core: one wave32 per (pixel, head); lane == channel.
// qkv layout per pixel row (768 f32): [q | k | v], each [head][32].
// Dot products via 5-step __shfl_xor butterfly; softmax replicated per lane.
// V rows are prefetched (global_prefetch_b8) during the QK pass.
// ---------------------------------------------------------------------------
__global__ void na_attn_core(const float* __restrict__ qkv, _Float16* __restrict__ out_h) {
    const int wave = (blockIdx.x * blockDim.x + threadIdx.x) >> 5;
    const int lane = threadIdx.x & 31;
    const int head = wave & (NA_NH - 1);
    const int pix  = wave >> 3;
    if (pix >= NA_PIX) return;

    const int h = pix / NA_W;
    const int w = pix - h * NA_W;
    int h0 = h - NA_KK / 2; h0 = h0 < 0 ? 0 : h0; h0 = h0 > NA_H - NA_KK ? NA_H - NA_KK : h0;
    int w0 = w - NA_KK / 2; w0 = w0 < 0 ? 0 : w0; w0 = w0 > NA_W - NA_KK ? NA_W - NA_KK : w0;

    const float scale = 0.17677669529663687f;   // 32^-0.5
    const float qv = qkv[(size_t)pix * 768 + head * NA_HD + lane] * scale;

    float attn[NA_WIN];
    float mx = -3.0e38f;
#pragma unroll
    for (int n = 0; n < NA_WIN; ++n) {
        const int nh = h0 + n / NA_KK;
        const int nw = w0 + n % NA_KK;
        const size_t base = (size_t)(nh * NA_W + nw) * 768 + head * NA_HD;
        __builtin_prefetch(qkv + base + 512 + lane, 0, 3);  // V row for the AV pass
        float p = qv * qkv[base + 256 + lane];              // K row
#pragma unroll
        for (int off = 16; off; off >>= 1) p += __shfl_xor(p, off, 32);
        attn[n] = p;
        mx = fmaxf(mx, p);
    }
    float s = 0.f;
#pragma unroll
    for (int n = 0; n < NA_WIN; ++n) { attn[n] = __expf(attn[n] - mx); s += attn[n]; }
    const float inv = 1.f / s;

    float acc = 0.f;
#pragma unroll
    for (int n = 0; n < NA_WIN; ++n) {
        const int nh = h0 + n / NA_KK;
        const int nw = w0 + n % NA_KK;
        const size_t vb = (size_t)(nh * NA_W + nw) * 768 + 512 + head * NA_HD;
        acc = fmaf(attn[n] * inv, qkv[vb + lane], acc);
    }
    out_h[(size_t)pix * NA_C + head * NA_HD + lane] = (_Float16)acc;
}

// ---------------------------------------------------------------------------
// Launch
// ---------------------------------------------------------------------------
extern "C" void kernel_launch(void* const* d_in, const int* in_sizes, int n_in,
                              void* d_out, int out_size, void* d_ws, size_t ws_size,
                              hipStream_t stream) {
    const float* x      = (const float*)d_in[0];   // [8192, 256]
    const float* w_qkv  = (const float*)d_in[1];   // [256, 768]
    const float* b_qkv  = (const float*)d_in[2];   // [768]
    const float* w_proj = (const float*)d_in[3];   // [256, 256]
    const float* b_proj = (const float*)d_in[4];   // [256]
    float* out = (float*)d_out;                    // [8192, 256]

    // Workspace carving (256B-aligned)
    char* ws = (char*)d_ws;
    const size_t N_X   = (size_t)NA_PIX * NA_C;        // 2,097,152
    const size_t N_WQ  = (size_t)NA_C * 3 * NA_C;      //   196,608
    const size_t N_WP  = (size_t)NA_C * NA_C;          //    65,536
    const size_t N_QKV = (size_t)NA_PIX * 3 * NA_C;    // 6,291,456

    _Float16* x_h     = (_Float16*)ws;                        size_t off = N_X * 2;
    _Float16* wqkv_t  = (_Float16*)(ws + off);                off += N_WQ * 2;
    _Float16* wproj_t = (_Float16*)(ws + off);                off += N_WP * 2;
    off = (off + 255) & ~(size_t)255;
    float*    qkv     = (float*)(ws + off);                   off += N_QKV * 4;
    _Float16* attn_h  = (_Float16*)(ws + off);                // N_X * 2 bytes

    // 1) convert inputs: x straight, weights transposed to [N,K] f16
    na_cvt_f16  <<<(int)((N_X  + 255) / 256), 256, 0, stream>>>(x, x_h, (int)N_X);
    na_cvt_f16_t<<<(int)((N_WQ + 255) / 256), 256, 0, stream>>>(w_qkv,  wqkv_t,  NA_C, 3 * NA_C);
    na_cvt_f16_t<<<(int)((N_WP + 255) / 256), 256, 0, stream>>>(w_proj, wproj_t, NA_C, NA_C);

    // 2) QKV projection: [8192,256] x [256,768] -> [8192,768]
    //    strips = 512 * 12 = 6144 waves; 8 waves per 256-thread block
    na_wmma_gemm<<<6144 / 8, 256, 0, stream>>>(x_h, wqkv_t, b_qkv, qkv, NA_PIX, 3 * NA_C);

    // 3) Neighborhood attention: 8192 * 8 waves
    na_attn_core<<<(NA_PIX * NA_NH) / 8, 256, 0, stream>>>(qkv, attn_h);

    // 4) Output projection: [8192,256] x [256,256] -> [8192,256]
    //    strips = 512 * 4 = 2048 waves
    na_wmma_gemm<<<2048 / 8, 256, 0, stream>>>(attn_h, wproj_t, b_proj, out, NA_PIX, NA_C);
}
